// STFeatureLearner_54640573939742
// MI455X (gfx1250) — matrix operand order repaired and checked by reading, not compiled, not run on hardware
//
#include <hip/hip_runtime.h>
#include <hip/hip_bf16.h>

#define B_ 8
#define T_ 12
#define N_ 1024
#define D_ 64
#define E_ 64
#define M_ 16

typedef __attribute__((ext_vector_type(16))) __bf16 v16bf;
typedef __attribute__((ext_vector_type(8)))  __bf16 v8bf;
typedef __attribute__((ext_vector_type(8)))  float  v8f;
typedef __attribute__((ext_vector_type(4)))  unsigned int v4u;
typedef __attribute__((ext_vector_type(8)))  int v8i;
typedef __attribute__((ext_vector_type(4)))  int v4i;

union Frag { v16bf v; v8bf h[2]; };

// ---------------------------------------------------------------------------
// TDM: issue a 2D tensor_load_to_lds of a [rows x 64] bf16 tile whose source
// rows stride by 1024 elements, depositing into LDS with a 16B pad after each
// 128B row (LDS row stride = 72 bf16), matching the WMMA fragment addressing.
// D# layout per CDNA5 ISA. Caller: one wave only.
// This toolchain exposes the 6-arg builtin (clang-23 / therock-10.0 form):
//   (uint32x4 g0, int32x8 g1, int32x4 g2, int32x4 g3, int32x8 g4, i32 cpol)
// ---------------------------------------------------------------------------
__device__ __forceinline__ void tdm_load_tile(const __bf16* gsrc,
                                              unsigned lds_byte_off,
                                              unsigned rows) {
  unsigned long long ga = (unsigned long long)(uintptr_t)gsrc;
  v4u g0;
  g0[0] = 1u;                                   // count=1 (valid), user mode
  g0[1] = lds_byte_off;                         // lds_addr
  g0[2] = (unsigned)(ga & 0xFFFFFFFFu);         // global_addr[31:0]
  g0[3] = (unsigned)((ga >> 32) & 0x1FFFFFFu)   // global_addr[56:32]
          | (2u << 30);                         // type=2 ("image")
  v8i g1;
  // data_size=1 (2B) <<16 | pad_enable<<20 | pad_interval=4 (32 DW)<<22 |
  // pad_amount=3 (4 DW = 16B)<<25
  g1[0] = (int)((1u << 16) | (1u << 20) | (4u << 22) | (3u << 25));
  g1[1] = (int)(((unsigned)N_ & 0xFFFFu) << 16);        // tensor_dim0 lo
  g1[2] = (int)(((unsigned)N_ & 0xFFFFu) << 16);        // dim0 hi=0 | dim1 lo
  g1[3] = (int)(((unsigned)64u) << 16);                 // dim1 hi=0 | tile_dim0=64
  g1[4] = (int)(rows & 0xFFFFu);                        // tile_dim1 | tile_dim2=0
  g1[5] = (int)N_;                                      // tensor_dim0_stride lo
  g1[6] = 0;                                            // stride0 hi | stride1 lo
  g1[7] = 0;                                            // stride1 hi
  v4i g2 = {0, 0, 0, 0};
  v4i g3 = {0, 0, 0, 0};
  v8i g4 = {0, 0, 0, 0, 0, 0, 0, 0};
  __builtin_amdgcn_tensor_load_to_lds(g0, g1, g2, g3, g4, 0);
}

// ---------------------------------------------------------------------------
// k0: phi_K = elu(PHI@W_K^T)+1 ; V = PHI@W_V^T ; KtV = phi_K^T @ V   (tiny)
// ---------------------------------------------------------------------------
__global__ __launch_bounds__(256) void k0_precompute(
    const float* __restrict__ PHI, const float* __restrict__ WK,
    const float* __restrict__ WV,
    float* __restrict__ phiK, float* __restrict__ Vm, float* __restrict__ KtV) {
  __shared__ float pk[M_ * E_];
  __shared__ float vv[M_ * D_];
  int tid = threadIdx.x;
  for (int idx = tid; idx < M_ * E_; idx += 256) {
    int m = idx / E_, e = idx % E_;
    float acc = 0.f;
    for (int j = 0; j < E_; ++j) acc += PHI[m * E_ + j] * WK[e * E_ + j];
    pk[idx] = acc > 0.f ? acc + 1.f : __expf(acc);   // elu(x)+1
  }
  for (int idx = tid; idx < M_ * D_; idx += 256) {
    int m = idx / D_, d = idx % D_;
    float acc = 0.f;
    for (int e = 0; e < E_; ++e) acc += PHI[m * E_ + e] * WV[d * E_ + e];
    vv[idx] = acc;
  }
  __syncthreads();
  for (int idx = tid; idx < M_ * E_; idx += 256) phiK[idx] = pk[idx];
  for (int idx = tid; idx < M_ * D_; idx += 256) Vm[idx] = vv[idx];
  for (int idx = tid; idx < E_ * D_; idx += 256) {
    int e = idx / D_, d = idx % D_;
    float acc = 0.f;
    for (int m = 0; m < M_; ++m) acc += pk[m * E_ + e] * vv[m * D_ + d];
    KtV[idx] = acc;
  }
}

// ---------------------------------------------------------------------------
// k0b: dinv[n] = (sum_m A_static[n,m] + 1e-12)^-0.5
// ---------------------------------------------------------------------------
__global__ __launch_bounds__(256) void k0b_dinv(
    const float* __restrict__ Ast, float* __restrict__ dinv) {
  __shared__ float red[256];
  int n = blockIdx.x;
  float acc = 0.f;
  for (int m = threadIdx.x; m < N_; m += 256) acc += Ast[n * N_ + m];
  red[threadIdx.x] = acc; __syncthreads();
  for (int s = 128; s > 0; s >>= 1) {
    if (threadIdx.x < s) red[threadIdx.x] += red[threadIdx.x + s];
    __syncthreads();
  }
  if (threadIdx.x == 0) dinv[n] = rsqrtf(red[0] + 1e-12f);
}

// ---------------------------------------------------------------------------
// k1: fused attention row kernel. Each 64-thread group owns (b,n), loops t,
// producing H_prime rows (fp32 + bf16 mirror) and the T-mean mv[b,n,:].
// ---------------------------------------------------------------------------
__global__ __launch_bounds__(256) void k1_attention_rows(
    const float* __restrict__ H, const float* __restrict__ WQ_g,
    const float* __restrict__ phiK_g, const float* __restrict__ Vm_g,
    const float* __restrict__ KtV_g, const float* __restrict__ outW_g,
    const float* __restrict__ out_b, const float* __restrict__ sma_g,
    const float* __restrict__ sma_b,
    float* __restrict__ Hprime, __bf16* __restrict__ Hprime16,
    float* __restrict__ mv) {
  __shared__ float WQ[E_ * 65];
  __shared__ float KT[E_ * 65];
  __shared__ float OW[D_ * 65];
  __shared__ float VS[M_ * 65];
  __shared__ float PK[M_ * 65];
  __shared__ float hb[4][64];
  __shared__ float pb[4][64];
  __shared__ float sb[4][M_];
  __shared__ float eb[4][M_];
  __shared__ float ab[4][M_];
  __shared__ float ob[4][64];
  int tid = threadIdx.x;
  for (int idx = tid; idx < 64 * 64; idx += 256) {
    int r = idx >> 6, c = idx & 63;
    WQ[r * 65 + c] = WQ_g[idx];
    KT[r * 65 + c] = KtV_g[idx];
    OW[r * 65 + c] = outW_g[idx];
  }
  for (int idx = tid; idx < M_ * 64; idx += 256) {
    int r = idx >> 6, c = idx & 63;
    VS[r * 65 + c] = Vm_g[idx];
    PK[r * 65 + c] = phiK_g[idx];
  }
  __syncthreads();
  int g = tid >> 6;
  int j = tid & 63;
  int gflat = blockIdx.x * 4 + g;
  int b = gflat >> 10;
  int n = gflat & (N_ - 1);
  float mvacc = 0.f;
  float obj = out_b[j], sgj = sma_g[j], sbj = sma_b[j];
  for (int t = 0; t < T_; ++t) {
    long r = ((long)(b * T_ + t) * N_ + n);
    hb[g][j] = H[r * 64 + j];
    __syncthreads();
    float q = 0.f;
#pragma unroll 8
    for (int d = 0; d < 64; ++d) q += hb[g][d] * WQ[j * 65 + d];
    pb[g][j] = q > 0.f ? q + 1.f : __expf(q);       // phi_Q
    __syncthreads();
    if (j < M_) {
      float s = 0.f;
#pragma unroll 8
      for (int e = 0; e < 64; ++e) s += PK[j * 65 + e] * pb[g][e];
      sb[g][j] = s * 0.125f;                         // / sqrt(E)
    }
    __syncthreads();
    if (j < M_) {
      float mx = sb[g][0];
      for (int m = 1; m < M_; ++m) mx = fmaxf(mx, sb[g][m]);
      eb[g][j] = __expf(sb[g][j] - mx);
    }
    __syncthreads();
    if (j < M_) {
      float sum = 0.f;
      for (int m = 0; m < M_; ++m) sum += eb[g][m];
      ab[g][j] = eb[g][j] / sum;
    }
    __syncthreads();
    float mn = 0.f;
#pragma unroll 8
    for (int e = 0; e < 64; ++e) mn += pb[g][e] * KT[e * 65 + j];
    float mem = 0.f;
#pragma unroll
    for (int m = 0; m < M_; ++m) mem += ab[g][m] * VS[m * 65 + j];
    mvacc += mem;
    float hpv = mn + hb[g][j];
    __syncthreads();
    pb[g][j] = hpv;                                  // reuse buffer for Hp
    __syncthreads();
    float o = obj;
#pragma unroll 8
    for (int d = 0; d < 64; ++d) o += pb[g][d] * OW[j * 65 + d];
    ob[g][j] = o;
    __syncthreads();
    float mu = 0.f;
#pragma unroll 8
    for (int d = 0; d < 64; ++d) mu += ob[g][d];
    mu *= (1.f / 64.f);
    float var = 0.f;
#pragma unroll 8
    for (int d = 0; d < 64; ++d) { float x = ob[g][d] - mu; var += x * x; }
    var *= (1.f / 64.f);
    float hv = (o - mu) * rsqrtf(var + 1e-5f) * sgj + sbj;
    Hprime[r * 64 + j] = hv;
    Hprime16[r * 64 + j] = (__bf16)hv;
    __syncthreads();
  }
  mv[((long)b * N_ + n) * 64 + j] = mvacc * (1.f / (float)T_);
}

// ---------------------------------------------------------------------------
// k2a: S[b,n,m] = dot(mv[b,n], mv[b,m]) / sqrt(D)  (fp32; output is exact)
// ---------------------------------------------------------------------------
__global__ __launch_bounds__(256) void k2a_scores(
    const float* __restrict__ mv, float* __restrict__ S) {
  __shared__ float rt[16][65];
  __shared__ float ct[16][65];
  int b = blockIdx.z;
  int rb = blockIdx.y * 16, cb = blockIdx.x * 16;
  int tid = threadIdx.x;
#pragma unroll
  for (int s = 0; s < 4; ++s) {
    int idx = tid + 256 * s;
    int rr = idx >> 6, cc = idx & 63;
    rt[rr][cc] = mv[((long)b * N_ + rb + rr) * 64 + cc];
    ct[rr][cc] = mv[((long)b * N_ + cb + rr) * 64 + cc];
  }
  __syncthreads();
  int ty = tid >> 4, tx = tid & 15;
  float acc = 0.f;
#pragma unroll 8
  for (int k = 0; k < 64; ++k) acc += rt[ty][k] * ct[tx][k];
  S[(long)b * N_ * N_ + (long)(rb + ty) * N_ + cb + tx] = acc * 0.125f;
}

// ---------------------------------------------------------------------------
// k2b: row softmax over N -> A_dynamic (in place, exact fp32); A_fused =
// bta*A_dyn + (1-bta)*dinv[n]*A_static*dinv[m] (exact fp32 + bf16 mirror)
// ---------------------------------------------------------------------------
__global__ __launch_bounds__(256) void k2b_softmax_fuse(
    float* __restrict__ Adyn, float* __restrict__ Afus,
    __bf16* __restrict__ Afus16,
    const float* __restrict__ Ast, const float* __restrict__ dinv,
    const float* __restrict__ beta_p) {
  __shared__ float red[256];
  int n = blockIdx.x, b = blockIdx.y, tid = threadIdx.x;
  float bta = fminf(fmaxf(beta_p[0], 0.f), 1.f);
  float* row = Adyn + (long)b * N_ * N_ + (long)n * N_;
  float mx = -3.4e38f;
  for (int m = tid; m < N_; m += 256) mx = fmaxf(mx, row[m]);
  red[tid] = mx; __syncthreads();
  for (int s = 128; s > 0; s >>= 1) {
    if (tid < s) red[tid] = fmaxf(red[tid], red[tid + s]);
    __syncthreads();
  }
  mx = red[0]; __syncthreads();
  float sum = 0.f;
  for (int m = tid; m < N_; m += 256) sum += __expf(row[m] - mx);
  red[tid] = sum; __syncthreads();
  for (int s = 128; s > 0; s >>= 1) {
    if (tid < s) red[tid] += red[tid + s];
    __syncthreads();
  }
  float inv = 1.f / red[0];
  float dn = dinv[n];
  float* frow = Afus + (long)b * N_ * N_ + (long)n * N_;
  __bf16* frow16 = Afus16 + (long)b * N_ * N_ + (long)n * N_;
  for (int m = tid; m < N_; m += 256) {
    float a = __expf(row[m] - mx) * inv;
    row[m] = a;
    float f = bta * a + (1.f - bta) * dn * Ast[n * N_ + m] * dinv[m];
    frow[m] = f;
    frow16[m] = (__bf16)f;
  }
}

// ---------------------------------------------------------------------------
// k3: diffusion hop  Xout[b,t] = A_fused[b] @ Xin[b,t]   via WMMA bf16.
// Block: 128 output rows x 64 cols (8 waves x (16x64)).
//  - A tile (128x64 bf16) staged by the Tensor Data Mover (tensor_load_to_lds,
//    one DMA per block issued by wave 0, pad descriptor -> LDS row stride 72).
//  - B tile staged transposed by VGPR path (v8bf loads, b16 scatter stores).
// Fragments are two contiguous 16B ds_load_b128 per lane, contiguous k order (16*hi + s)
// for both A and B. fp32 accumulate; optional bf16 mirror of the output.
// ---------------------------------------------------------------------------
__global__ __launch_bounds__(256) void k3_hop_wmma(
    const __bf16* __restrict__ Afus16, const __bf16* __restrict__ Xin16,
    float* __restrict__ Xout, __bf16* __restrict__ Xout16) {
  __shared__ __align__(16) __bf16 Alds[128 * 72];
  __shared__ __align__(16) __bf16 Blds[64 * 72];
  int tid = threadIdx.x;
  int bt = blockIdx.y;            // 0..B*T-1
  int b = bt / T_;
  int r0 = blockIdx.x * 128;
  const __bf16* Ab = Afus16 + (long)b * N_ * N_;
  const __bf16* Xb = Xin16 + (long)bt * N_ * 64;
  float* Ob = Xout + (long)bt * N_ * 64;
  int wave = tid >> 5;
  int lane = tid & 31;
  int lt = lane & 15;
  int hi = lane >> 4;
  unsigned alds_off = (unsigned)(uintptr_t)(void*)&Alds[0];
  v8f c0 = {}, c1 = {}, c2 = {}, c3 = {};
  for (int kt = 0; kt < N_; kt += 64) {
    // --- stage A tile via TDM (wave 0 only; TDM ignores EXEC) ---
    if (tid < 32) {
      tdm_load_tile(Ab + (long)r0 * N_ + kt, alds_off, 128u);
      __builtin_amdgcn_s_wait_tensorcnt(0);
    }
    // --- stage B tile transposed: Blds[n][k] = Xin16[kt+k][n] ---
#pragma unroll
    for (int s = 0; s < 2; ++s) {
      int q = tid + 256 * s;                 // v8bf chunk index, 0..511
      int kr = q >> 3;                       // k row within tile
      int c8 = (q & 7) << 3;                 // first of 8 consecutive n
      v8bf v = *(const v8bf*)(Xb + (long)(kt + kr) * 64 + c8);
#pragma unroll
      for (int i = 0; i < 8; ++i) Blds[(c8 + i) * 72 + kr] = v[i];
    }
    __syncthreads();
#pragma unroll
    for (int ks = 0; ks < 64; ks += 32) {
      // load all fragments first, then issue the 4 WMMAs back-to-back
      Frag a;
      const __bf16* ap = &Alds[(wave * 16 + lt) * 72 + ks + (hi ? 16 : 0)];
      a.h[0] = *(const v8bf*)(ap);
      a.h[1] = *(const v8bf*)(ap + 8);
      Frag b0, b1, b2, b3;
      const __bf16* bp0 = &Blds[(0 * 16 + lt) * 72 + ks + (hi ? 16 : 0)];
      const __bf16* bp1 = &Blds[(1 * 16 + lt) * 72 + ks + (hi ? 16 : 0)];
      const __bf16* bp2 = &Blds[(2 * 16 + lt) * 72 + ks + (hi ? 16 : 0)];
      const __bf16* bp3 = &Blds[(3 * 16 + lt) * 72 + ks + (hi ? 16 : 0)];
      b0.h[0] = *(const v8bf*)(bp0); b0.h[1] = *(const v8bf*)(bp0 + 8);
      b1.h[0] = *(const v8bf*)(bp1); b1.h[1] = *(const v8bf*)(bp1 + 8);
      b2.h[0] = *(const v8bf*)(bp2); b2.h[1] = *(const v8bf*)(bp2 + 8);
      b3.h[0] = *(const v8bf*)(bp3); b3.h[1] = *(const v8bf*)(bp3 + 8);
      c0 = __builtin_amdgcn_wmma_f32_16x16x32_bf16(false, a.v, false, b0.v,
                                                   (short)0, c0, false, false);
      c1 = __builtin_amdgcn_wmma_f32_16x16x32_bf16(false, a.v, false, b1.v,
                                                   (short)0, c1, false, false);
      c2 = __builtin_amdgcn_wmma_f32_16x16x32_bf16(false, a.v, false, b2.v,
                                                   (short)0, c2, false, false);
      c3 = __builtin_amdgcn_wmma_f32_16x16x32_bf16(false, a.v, false, b3.v,
                                                   (short)0, c3, false, false);
    }
    __syncthreads();
  }
  // epilogue: C layout VGPR k -> row k (+8 for lanes 16..31), col = lane&15
  int orow = r0 + wave * 16 + hi * 8;
#pragma unroll
  for (int k = 0; k < 8; ++k) {
    Ob[(long)(orow + k) * 64 + 0 * 16 + lt] = c0[k];
    Ob[(long)(orow + k) * 64 + 1 * 16 + lt] = c1[k];
    Ob[(long)(orow + k) * 64 + 2 * 16 + lt] = c2[k];
    Ob[(long)(orow + k) * 64 + 3 * 16 + lt] = c3[k];
  }
  if (Xout16) {
    __bf16* Ob16 = Xout16 + (long)bt * N_ * 64;
#pragma unroll
    for (int k = 0; k < 8; ++k) {
      Ob16[(long)(orow + k) * 64 + 0 * 16 + lt] = (__bf16)c0[k];
      Ob16[(long)(orow + k) * 64 + 1 * 16 + lt] = (__bf16)c1[k];
      Ob16[(long)(orow + k) * 64 + 2 * 16 + lt] = (__bf16)c2[k];
      Ob16[(long)(orow + k) * 64 + 3 * 16 + lt] = (__bf16)c3[k];
    }
  }
}

// ---------------------------------------------------------------------------
// k4: out = LN( [H',X1,X2] @ proj_W^T + proj_b )  (fp32 row kernel)
// ---------------------------------------------------------------------------
__global__ __launch_bounds__(256) void k4_proj_ln(
    const float* __restrict__ Hp, const float* __restrict__ X1,
    const float* __restrict__ X2, const float* __restrict__ PW_g,
    const float* __restrict__ pb_g, const float* __restrict__ agg_g,
    const float* __restrict__ agg_b, float* __restrict__ Hout) {
  __shared__ float PW[64 * 193];
  __shared__ float cat[4][192];
  __shared__ float ob[4][64];
  int tid = threadIdx.x;
  for (int idx = tid; idx < 64 * 192; idx += 256) {
    int o = idx / 192, c = idx % 192;
    PW[o * 193 + c] = PW_g[idx];
  }
  __syncthreads();
  int g = tid >> 6, j = tid & 63;
  long r = (long)blockIdx.x * 4 + g;
  cat[g][j]       = Hp[r * 64 + j];
  cat[g][64 + j]  = X1[r * 64 + j];
  cat[g][128 + j] = X2[r * 64 + j];
  __syncthreads();
  float o = pb_g[j];
#pragma unroll 8
  for (int c = 0; c < 192; ++c) o += cat[g][c] * PW[j * 193 + c];
  ob[g][j] = o;
  __syncthreads();
  float mu = 0.f;
#pragma unroll 8
  for (int d = 0; d < 64; ++d) mu += ob[g][d];
  mu *= (1.f / 64.f);
  float var = 0.f;
#pragma unroll 8
  for (int d = 0; d < 64; ++d) { float x = ob[g][d] - mu; var += x * x; }
  var *= (1.f / 64.f);
  Hout[r * 64 + j] = (o - mu) * rsqrtf(var + 1e-5f) * agg_g[j] + agg_b[j];
}

// ---------------------------------------------------------------------------
extern "C" void kernel_launch(void* const* d_in, const int* in_sizes, int n_in,
                              void* d_out, int out_size, void* d_ws,
                              size_t ws_size, hipStream_t stream) {
  const float* H_temp   = (const float*)d_in[0];
  const float* A_static = (const float*)d_in[1];
  const float* W_Q      = (const float*)d_in[2];
  const float* W_K      = (const float*)d_in[3];
  const float* W_V      = (const float*)d_in[4];
  const float* PHI      = (const float*)d_in[5];
  const float* out_W    = (const float*)d_in[6];
  const float* out_b    = (const float*)d_in[7];
  const float* sma_g    = (const float*)d_in[8];
  const float* sma_b    = (const float*)d_in[9];
  const float* beta     = (const float*)d_in[10];
  const float* proj_W   = (const float*)d_in[11];
  const float* proj_b   = (const float*)d_in[12];
  const float* agg_g    = (const float*)d_in[13];
  const float* agg_b    = (const float*)d_in[14];

  float* out  = (float*)d_out;
  float* Hsp  = out;                                  // [B,T,N,D]
  float* Adyn = out + 6291456;                        // [B,N,N]
  float* Afus = out + 6291456 + 8388608;              // [B,N,N]

  float* ws    = (float*)d_ws;                        // ~120 MB used
  float* phiK  = ws;                                  // 16*64
  float* Vm    = ws + 1024;                           // 16*64
  float* KtV   = ws + 2048;                           // 64*64
  float* dinv  = ws + 6144;                           // 1024
  float* mv    = ws + 7168;                           // B*N*64
  float* Hpr   = ws + 531456;                         // R*64 fp32
  float* X1    = ws + 6822912;                        // R*64 fp32
  float* X2    = ws + 13114368;                       // R*64 fp32
  __bf16* Af16 = (__bf16*)(ws + 19405824);            // B*N*N bf16
  __bf16* Hp16 = (__bf16*)(ws + 23600128);            // R*64 bf16
  __bf16* X116 = (__bf16*)(ws + 26745856);            // R*64 bf16

  k0_precompute<<<1, 256, 0, stream>>>(PHI, W_K, W_V, phiK, Vm, KtV);
  k0b_dinv<<<N_, 256, 0, stream>>>(A_static, dinv);
  k1_attention_rows<<<2048, 256, 0, stream>>>(H_temp, W_Q, phiK, Vm, KtV,
                                              out_W, out_b, sma_g, sma_b,
                                              Hpr, Hp16, mv);
  k2a_scores<<<dim3(64, 64, 8), 256, 0, stream>>>(mv, Adyn);
  k2b_softmax_fuse<<<dim3(N_, B_), 256, 0, stream>>>(Adyn, Afus, Af16,
                                                     A_static, dinv, beta);
  k3_hop_wmma<<<dim3(8, B_ * T_), 256, 0, stream>>>(Af16, Hp16, X1, X116);
  k3_hop_wmma<<<dim3(8, B_ * T_), 256, 0, stream>>>(Af16, X116, X2, nullptr);
  k4_proj_ln<<<24576, 256, 0, stream>>>(Hpr, X1, X2, proj_W, proj_b,
                                        agg_g, agg_b, Hsp);
}
